// MultiHeadAttention_36764920054077
// MI455X (gfx1250) — matrix operand order repaired and checked by reading, not compiled, not run on hardware
//
#include <hip/hip_runtime.h>

// ---------------------------------------------------------------------------
// MHA forward for MI455X (gfx1250, wave32, WMMA 16x16x32 f16->f32 accum)
// Shapes fixed by the reference: B=4, S=2048, E=768, H=12, D=64.
// Reference bug reproduced: v is projected with Wk/bk (Wv, bv unused).
// Round 2: stage Q/V (attention) and A (GEMM) tiles via the Tensor Data
// Mover (TENSOR_LOAD_TO_LDS + s_wait_tensorcnt), using D# pad fields to
// keep the bank-conflict-avoiding padded LDS rows.
// ---------------------------------------------------------------------------

typedef __attribute__((ext_vector_type(16))) _Float16     v16h;
typedef __attribute__((ext_vector_type(8)))  float        v8f;
typedef __attribute__((ext_vector_type(4)))  unsigned int u32x4;
typedef __attribute__((ext_vector_type(4)))  int          i32x4;
typedef __attribute__((ext_vector_type(8)))  int          i32x8;
typedef __attribute__((ext_vector_type(4)))  float        f32x4;
typedef __attribute__((ext_vector_type(4)))  _Float16     h16x4;

#define EMBD   768
#define HEADS  12
#define HDIM   64
#define BATCH  4
#define SEQ    2048
#define MROWS  (BATCH * SEQ)   // 8192

#if defined(__HIP_DEVICE_COMPILE__)
#if __has_builtin(__builtin_amdgcn_tensor_load_to_lds)
#define HAVE_TDM 1
#if defined(__clang_major__) && (__clang_major__ >= 23)
#define TDM_LOAD(g0, g1, g2, g3) \
  __builtin_amdgcn_tensor_load_to_lds((g0), (g1), (g2), (g3), (i32x8)0, 0)
#else
#define TDM_LOAD(g0, g1, g2, g3) \
  __builtin_amdgcn_tensor_load_to_lds((g0), (g1), (g2), (g3), 0)
#endif
#endif
#endif

union FragU { v16h v; u32x4 q[2]; };

// A-matrix 16x32 f16 fragment from LDS tile (row-major, `stride` halves).
// Layout (ISA 7.12.2): lanes 0-15 row M=lane, halves K=[0..7],[16..23];
// lanes 16-31 row M=lane-16, halves K=[8..15],[24..31].
__device__ __forceinline__ v16h ld_frag_a(const _Float16* base, int stride, int lane) {
  const int r  = lane & 15;
  const int hs = (lane >> 4) & 1;
  const _Float16* p = base + r * stride + hs * 8;
  FragU f;
  f.q[0] = *reinterpret_cast<const u32x4*>(p);
  f.q[1] = *reinterpret_cast<const u32x4*>(p + 16);
  return f.v;
}

// B-matrix 32x16 f16 fragment from LDS tile stored [k][n] (`stride` halves).
// Lane column N=lane&15; K=[0..7],[16..23] (lanes 0-15) / [8..15],[24..31] (lanes 16-31).
__device__ __forceinline__ v16h ld_frag_b(const _Float16* base, int stride, int lane) {
  const int n  = lane & 15;
  const int hs = (lane >> 4) & 1;
  v16h f;
#pragma unroll
  for (int s = 0; s < 8; ++s) {
    f[s]     = base[(hs * 8 + s) * stride + n];
    f[s + 8] = base[(16 + hs * 8 + s) * stride + n];
  }
  return f;
}

__device__ __forceinline__ v8f wmma_f16(v16h a, v16h b, v8f c) {
  return __builtin_amdgcn_wmma_f32_16x16x32_f16(false, a, false, b, (short)0, c,
                                                false, false);
}

__device__ __forceinline__ v8f zero_v8f() {
  v8f z;
#pragma unroll
  for (int e = 0; e < 8; ++e) z[e] = 0.0f;
  return z;
}

#ifdef HAVE_TDM
// Low 32 bits of the flat LDS-aperture address are the LDS byte offset
// (ISA 10.2).
__device__ __forceinline__ unsigned lds_off(const void* p) {
  return (unsigned)(uintptr_t)p;
}

// Issue a 2-D TENSOR_LOAD_TO_LDS. All sizes/strides in 2-byte elements.
// pad_icode: LDS pad interval code (0=2 DW,1=4,2=8,3=16,4=32,...)
// pad_acode: LDS pad amount code   (0=1 DW,1=2,2=3,3=4,...)
__device__ __forceinline__ void tdm_load_2d(unsigned lds_byte, const void* gptr,
                                            unsigned td0, unsigned td1,
                                            unsigned stride0, unsigned tile0,
                                            unsigned tile1, unsigned pad_icode,
                                            unsigned pad_acode) {
  unsigned long long ga = (unsigned long long)(uintptr_t)gptr;
  u32x4 g0;
  g0[0] = 1u;                                   // count=1, user descriptor
  g0[1] = lds_byte;                             // lds_addr
  g0[2] = (unsigned)(ga & 0xFFFFFFFFull);       // global_addr[31:0]
  g0[3] = (unsigned)((ga >> 32) & 0x01FFFFFFull) | (2u << 30);  // type=2
  i32x8 g1;
  g1[0] = (int)((1u << 16) |                    // data_size = 2 bytes
                (1u << 20) |                    // pad_enable
                (pad_icode << 22) | (pad_acode << 25));
  g1[1] = (int)((td0 & 0xFFFFu) << 16);         // tensor_dim0[15:0] @ bit48
  g1[2] = (int)(((td0 >> 16) & 0xFFFFu) | ((td1 & 0xFFFFu) << 16));
  g1[3] = (int)(((td1 >> 16) & 0xFFFFu) | ((tile0 & 0xFFFFu) << 16));
  g1[4] = (int)(tile1 & 0xFFFFu);               // tile_dim1 (tile_dim2=0)
  g1[5] = (int)stride0;                         // tensor_dim0_stride[31:0]
  g1[6] = 0;                                    // stride0 hi / stride1 lo
  g1[7] = 0;
  i32x4 zz = (i32x4)0;
  TDM_LOAD(g0, g1, zz, zz);
}
#endif

// ---------------------------------------------------------------------------
// f32 -> f16 conversion (vectorized), halves GEMM-side HBM traffic.
// ---------------------------------------------------------------------------
__global__ void cvt_f32_f16(const float* __restrict__ in,
                            _Float16* __restrict__ out, int n4) {
  int i  = blockIdx.x * blockDim.x + threadIdx.x;
  int st = gridDim.x * blockDim.x;
  for (; i < n4; i += st) {
    f32x4 v = reinterpret_cast<const f32x4*>(in)[i];
    h16x4 h;
    h.x = (_Float16)v.x; h.y = (_Float16)v.y;
    h.z = (_Float16)v.z; h.w = (_Float16)v.w;
    reinterpret_cast<h16x4*>(out)[i] = h;
  }
}

// ---------------------------------------------------------------------------
// GEMM: out[m,n] = sum_k X[m,k] * W[n,k] + bias[n]  (X:[8192x768], W:[768x768])
// OUT_MODE 0: f16 head-major [b][h][s][d] (QKV projections)
// OUT_MODE 1: f32 row-major [m][768]      (final output projection)
// Block tile 128x128, BK=32; 8 waves, each owns a 64x32 slab (4x2 WMMA tiles).
// A tile staged by TDM (wave 0); W tile transposed into LDS by all waves.
// ---------------------------------------------------------------------------
template <int OUT_MODE>
__launch_bounds__(256)
__global__ void gemm_wmma_768(const _Float16* __restrict__ X,
                              const _Float16* __restrict__ W,
                              const float* __restrict__ bias,
                              void* __restrict__ out) {
  constexpr int K = EMBD, BM = 128, BN = 128, BK = 32;
  __shared__ _Float16 sA[BM][BK + 8];   // [m][k], padded rows (32+8 halves)
  __shared__ _Float16 sB[BK][BN + 8];   // W transposed -> [k][n], padded rows

  const int tid  = threadIdx.x;
  const int lane = tid & 31;
  const int wid  = tid >> 5;
  const int wm   = wid & 1;       // 0..1 -> 64-row slab
  const int wn   = wid >> 1;      // 0..3 -> 32-col slab
  const int m0   = blockIdx.y * BM;
  const int n0   = blockIdx.x * BN;

  v8f acc[4][2];
#pragma unroll
  for (int i = 0; i < 4; ++i)
#pragma unroll
    for (int j = 0; j < 2; ++j) acc[i][j] = zero_v8f();

  for (int kt = 0; kt < K; kt += BK) {
#ifdef HAVE_TDM
    // ---- stage A tile via Tensor Data Mover: 128 rows x 32 halves,
    //      LDS rows padded 32+8 halves (interval 16 DW, pad 4 DW) ----
    if (wid == 0)
      tdm_load_2d(lds_off(&sA[0][0]), X + (size_t)m0 * K + kt,
                  /*td0=*/K, /*td1=*/MROWS - m0, /*stride0=*/K,
                  /*tile0=*/BK, /*tile1=*/BM, /*pad_i=*/3, /*pad_a=*/3);
#else
    for (int t = tid; t < (BM * BK) / 8; t += 256) {
      int row = t >> 2;
      int ck  = (t & 3) << 3;
      const _Float16* src = X + (size_t)(m0 + row) * K + kt + ck;
      *reinterpret_cast<u32x4*>(&sA[row][ck]) =
          *reinterpret_cast<const u32x4*>(src);
      if (kt + BK < K) __builtin_prefetch(src + BK, 0, 1);
    }
#endif
    // ---- stage W tile transposed: global [n][k] -> LDS [k][n] ----
    for (int t = tid; t < (BN * BK) / 8; t += 256) {
      int nr = t >> 2;
      int ck = (t & 3) << 3;
      const _Float16* src = W + (size_t)(n0 + nr) * K + kt + ck;
      u32x4 raw = *reinterpret_cast<const u32x4*>(src);
      const _Float16* hv = reinterpret_cast<const _Float16*>(&raw);
#pragma unroll
      for (int i = 0; i < 8; ++i) sB[ck + i][nr] = hv[i];
      if (kt + BK < K) __builtin_prefetch(src + BK, 0, 1);
    }
#ifdef HAVE_TDM
#if __has_builtin(__builtin_amdgcn_s_wait_tensorcnt)
    if (wid == 0) __builtin_amdgcn_s_wait_tensorcnt(0);
#endif
#endif
    __syncthreads();

    v16h bf[2];
#pragma unroll
    for (int j = 0; j < 2; ++j)
      bf[j] = ld_frag_b(&sB[0][wn * 32 + j * 16], BN + 8, lane);
#pragma unroll
    for (int i = 0; i < 4; ++i) {
      v16h af = ld_frag_a(&sA[wm * 64 + i * 16][0], BK + 8, lane);
#pragma unroll
      for (int j = 0; j < 2; ++j) acc[i][j] = wmma_f16(af, bf[j], acc[i][j]);
    }
    __syncthreads();
  }

  // ---- epilogue: C layout (7.12.2): VGPR e -> M = e + 8*(lane>=16), N = lane&15
  const int hs8 = ((lane >> 4) & 1) << 3;
#pragma unroll
  for (int i = 0; i < 4; ++i) {
#pragma unroll
    for (int j = 0; j < 2; ++j) {
      const int n  = n0 + wn * 32 + j * 16 + (lane & 15);
      const float bv = bias[n];
#pragma unroll
      for (int e = 0; e < 8; ++e) {
        const int m   = m0 + wm * 64 + i * 16 + e + hs8;
        const float r = acc[i][j][e] + bv;
        if (OUT_MODE == 0) {
          const int b = m >> 11, s = m & (SEQ - 1);
          const int h = n >> 6,  d = n & (HDIM - 1);
          reinterpret_cast<_Float16*>(out)
              [(((size_t)(b * HEADS + h) * SEQ + s) << 6) + d] = (_Float16)r;
        } else {
          reinterpret_cast<float*>(out)[(size_t)m * EMBD + n] = r;
        }
      }
    }
  }
}

// ---------------------------------------------------------------------------
// Flash attention: one block per (b, h, 64-query tile); streams 32 key tiles
// of 64 with online softmax. Q and V tiles staged by TDM; K tile transposed
// into LDS manually (TDM has no transpose).
// ---------------------------------------------------------------------------
__launch_bounds__(256)
__global__ void flash_attn(const _Float16* __restrict__ Qh,
                           const _Float16* __restrict__ Kh,
                           const _Float16* __restrict__ Vh,
                           _Float16* __restrict__ Y) {
  __shared__ _Float16 sQ [64][72];   // [q][d]
  __shared__ _Float16 sKT[64][72];   // [d][key]  (transposed at staging)
  __shared__ _Float16 sV [64][72];   // [key][d]  (already B-layout)
  __shared__ _Float16 sP [64][72];   // softmaxed probs, f16
  __shared__ float    sS [64][68];   // raw scores, f32
  __shared__ float    sMx[64], sLs[64], sAl[64];

  const int tid  = threadIdx.x;
  const int lane = tid & 31;
  const int wid  = tid >> 5;
  const int iq   = wid & 3;          // query sub-tile (16 rows)
  const int jp   = (wid >> 2) << 1;  // column sub-tile pair base
  const int qt   = blockIdx.x;
  const int bh   = blockIdx.y;
  const int b    = bh / HEADS, h = bh % HEADS;
  const int hs8  = ((lane >> 4) & 1) << 3;

  const _Float16* Qb = Qh + (size_t)bh * SEQ * HDIM;
  const _Float16* Kb = Kh + (size_t)bh * SEQ * HDIM;
  const _Float16* Vb = Vh + (size_t)bh * SEQ * HDIM;

#ifdef HAVE_TDM
  // 64x64-half tile, LDS rows padded 64+8 halves (interval 32 DW, pad 4 DW).
  if (wid == 0)
    tdm_load_2d(lds_off(&sQ[0][0]), Qb + (size_t)qt * 64 * HDIM,
                /*td0=*/HDIM, /*td1=*/SEQ, /*stride0=*/HDIM,
                /*tile0=*/HDIM, /*tile1=*/64, /*pad_i=*/4, /*pad_a=*/3);
#else
  for (int t = tid; t < (64 * 64) / 8; t += 256) {
    int r = t >> 3, c = (t & 7) << 3;
    *reinterpret_cast<u32x4*>(&sQ[r][c]) =
        *reinterpret_cast<const u32x4*>(Qb + (size_t)(qt * 64 + r) * HDIM + c);
  }
#endif
  if (tid < 64) { sMx[tid] = -3.402823466e38f; sLs[tid] = 0.0f; }

  v8f oacc[2];
  oacc[0] = zero_v8f();
  oacc[1] = zero_v8f();
  const float inv_scale = 0.125f;  // 1/sqrt(64)

  for (int kt = 0; kt < SEQ / 64; ++kt) {
    __syncthreads();  // previous P*V reads of sKT/sV/sP complete
#ifdef HAVE_TDM
    if (wid == 0)
      tdm_load_2d(lds_off(&sV[0][0]), Vb + (size_t)kt * 64 * HDIM,
                  /*td0=*/HDIM, /*td1=*/SEQ, /*stride0=*/HDIM,
                  /*tile0=*/HDIM, /*tile1=*/64, /*pad_i=*/4, /*pad_a=*/3);
#endif
    // ---- stage K tile transposed: global [key][d] -> LDS [d][key] ----
    for (int t = tid; t < (64 * 64) / 8; t += 256) {
      int r = t >> 3, c = (t & 7) << 3;
      u32x4 kraw = *reinterpret_cast<const u32x4*>(
          Kb + (size_t)(kt * 64 + r) * HDIM + c);
      const _Float16* kv = reinterpret_cast<const _Float16*>(&kraw);
#pragma unroll
      for (int i = 0; i < 8; ++i) sKT[c + i][r] = kv[i];
#ifndef HAVE_TDM
      *reinterpret_cast<u32x4*>(&sV[r][c]) = *reinterpret_cast<const u32x4*>(
          Vb + (size_t)(kt * 64 + r) * HDIM + c);
#endif
    }
#ifdef HAVE_TDM
#if __has_builtin(__builtin_amdgcn_s_wait_tensorcnt)
    if (wid == 0) __builtin_amdgcn_s_wait_tensorcnt(0);  // covers Q (iter 0) + V
#endif
#endif
    __syncthreads();

    // ---- scores = Q * K^T (each wave: 2 of the 16 16x16 tiles) ----
#pragma unroll
    for (int jj = 0; jj < 2; ++jj) {
      v8f sc = zero_v8f();
#pragma unroll
      for (int kd = 0; kd < 2; ++kd) {
        v16h qa = ld_frag_a(&sQ[iq * 16][kd * 32], 72, lane);
        v16h kb = ld_frag_b(&sKT[kd * 32][(jp + jj) * 16], 72, lane);
        sc = wmma_f16(qa, kb, sc);
      }
      const int n = (jp + jj) * 16 + (lane & 15);
#pragma unroll
      for (int e = 0; e < 8; ++e) sS[iq * 16 + e + hs8][n] = sc[e] * inv_scale;
    }
    __syncthreads();

    // ---- online softmax: 4 threads per query row ----
    {
      const int row = tid >> 2;
      const int sub = tid & 3;
      float mt = -3.402823466e38f;
#pragma unroll
      for (int c = 0; c < 16; ++c) mt = fmaxf(mt, sS[row][sub * 16 + c]);
      mt = fmaxf(mt, __shfl_xor(mt, 1, 4));
      mt = fmaxf(mt, __shfl_xor(mt, 2, 4));
      const float m_old = sMx[row];
      const float m_new = fmaxf(m_old, mt);
      float ps = 0.0f;
#pragma unroll
      for (int c = 0; c < 16; ++c) {
        float pv = __expf(sS[row][sub * 16 + c] - m_new);
        ps += pv;
        sP[row][sub * 16 + c] = (_Float16)pv;
      }
      ps += __shfl_xor(ps, 1, 4);
      ps += __shfl_xor(ps, 2, 4);
      if (sub == 0) {
        const float alpha = __expf(m_old - m_new);
        sAl[row] = alpha;
        sLs[row] = alpha * sLs[row] + ps;
        sMx[row] = m_new;
      }
    }
    __syncthreads();

    // ---- rescale running O, then O += P * V ----
#pragma unroll
    for (int jj = 0; jj < 2; ++jj)
#pragma unroll
      for (int e = 0; e < 8; ++e) oacc[jj][e] *= sAl[iq * 16 + e + hs8];
#pragma unroll
    for (int kp = 0; kp < 2; ++kp) {
      v16h pa = ld_frag_a(&sP[iq * 16][kp * 32], 72, lane);
#pragma unroll
      for (int jj = 0; jj < 2; ++jj) {
        v16h vb = ld_frag_b(&sV[kp * 32][(jp + jj) * 16], 72, lane);
        oacc[jj] = wmma_f16(pa, vb, oacc[jj]);
      }
    }
  }
  __syncthreads();

  // ---- normalize and store y as f16 row-major [b][s][E] for final GEMM ----
#pragma unroll
  for (int jj = 0; jj < 2; ++jj) {
    const int dcol = (jp + jj) * 16 + (lane & 15);
#pragma unroll
    for (int e = 0; e < 8; ++e) {
      const int r  = iq * 16 + e + hs8;
      const int sg = qt * 64 + r;
      const float val = oacc[jj][e] / sLs[r];
      Y[(size_t)(b * SEQ + sg) * EMBD + h * HDIM + dcol] = (_Float16)val;
    }
  }
}

// ---------------------------------------------------------------------------
// Host launch
// ---------------------------------------------------------------------------
extern "C" void kernel_launch(void* const* d_in, const int* in_sizes, int n_in,
                              void* d_out, int out_size, void* d_ws,
                              size_t ws_size, hipStream_t stream) {
  (void)in_sizes; (void)n_in; (void)out_size; (void)ws_size;
  const float* q  = (const float*)d_in[0];
  const float* k  = (const float*)d_in[1];
  const float* v  = (const float*)d_in[2];
  const float* Wq = (const float*)d_in[3];
  const float* bq = (const float*)d_in[4];
  const float* Wk = (const float*)d_in[5];
  const float* bk = (const float*)d_in[6];
  // d_in[7]=Wv, d_in[8]=bv intentionally unused: reference projects v with Wk/bk.
  const float* Wo = (const float*)d_in[9];
  const float* bo = (const float*)d_in[10];

  const size_t MBS = (size_t)MROWS * EMBD;   // 6,291,456 elements
  const size_t WSZ = (size_t)EMBD * EMBD;    //   589,824 elements
  _Float16* p   = (_Float16*)d_ws;
  _Float16* qh  = p; p += MBS;   // f16 inputs
  _Float16* kh  = p; p += MBS;
  _Float16* vh  = p; p += MBS;
  _Float16* wqh = p; p += WSZ;   // f16 weights
  _Float16* wkh = p; p += WSZ;
  _Float16* woh = p; p += WSZ;
  _Float16* qpd = p; p += MBS;   // head-major projections
  _Float16* kpd = p; p += MBS;
  _Float16* vpd = p; p += MBS;
  _Float16* yh  = p; p += MBS;   // attention output, row-major

  cvt_f32_f16<<<2048, 256, 0, stream>>>(q, qh, (int)(MBS / 4));
  cvt_f32_f16<<<2048, 256, 0, stream>>>(k, kh, (int)(MBS / 4));
  cvt_f32_f16<<<2048, 256, 0, stream>>>(v, vh, (int)(MBS / 4));
  cvt_f32_f16<<<288, 256, 0, stream>>>(Wq, wqh, (int)(WSZ / 4));
  cvt_f32_f16<<<288, 256, 0, stream>>>(Wk, wkh, (int)(WSZ / 4));
  cvt_f32_f16<<<288, 256, 0, stream>>>(Wo, woh, (int)(WSZ / 4));

  dim3 gp(EMBD / 128, MROWS / 128);  // (6, 64)
  gemm_wmma_768<0><<<gp, 256, 0, stream>>>(qh, wqh, bq, qpd);
  gemm_wmma_768<0><<<gp, 256, 0, stream>>>(kh, wkh, bk, kpd);
  gemm_wmma_768<0><<<gp, 256, 0, stream>>>(vh, wkh, bk, vpd);  // bug-faithful
  flash_attn<<<dim3(SEQ / 64, BATCH * HEADS), 256, 0, stream>>>(qpd, kpd, vpd, yh);
  gemm_wmma_768<1><<<gp, 256, 0, stream>>>(yh, woh, bo, d_out);
}
